// MultiheadAttention_24910810317058
// MI455X (gfx1250) — matrix operand order repaired and checked by reading, not compiled, not run on hardware
//
#include <hip/hip_runtime.h>
#include <cstdint>

#define D_MODEL 1024
#define NHEAD   16
#define DHEAD   64
#define SEQ     2048
#define BATCH   2
#define MTOT    (BATCH*SEQ)   /* 4096 */

typedef _Float16 h4  __attribute__((ext_vector_type(4)));
typedef _Float16 h8  __attribute__((ext_vector_type(8)));
typedef _Float16 h16 __attribute__((ext_vector_type(16)));
typedef float    f4  __attribute__((ext_vector_type(4)));
typedef float    f8  __attribute__((ext_vector_type(8)));
typedef unsigned int u32x4 __attribute__((ext_vector_type(4)));
typedef int          i32x4 __attribute__((ext_vector_type(4)));
typedef int          i32x8 __attribute__((ext_vector_type(8)));

static __device__ __forceinline__ h16 cat16(h8 lo, h8 hi) {
    h16 r;
#pragma unroll
    for (int i = 0; i < 8; ++i) { r[i] = lo[i]; r[i + 8] = hi[i]; }
    return r;
}

// Load a 16-half fragment as two 16-byte chunks (p0 -> elems 0..7, p1 -> 8..15)
static __device__ __forceinline__ h16 ld2(const _Float16* p0, const _Float16* p1) {
    return cat16(*(const h8*)p0, *(const h8*)p1);
}

static __device__ __forceinline__ f8 wmma_f16(h16 a, h16 b, f8 c) {
    return __builtin_amdgcn_wmma_f32_16x16x32_f16(
        /*neg_a=*/false, a, /*neg_b=*/false, b,
        /*c_mod=*/(short)0, c, /*reuse_a=*/false, /*reuse_b=*/false);
}

// ---------------------------------------------------------------------------
// Tensor Data Mover: 2D tile (tile_dim0=32 halves x tile_dim1=128 rows) from
// global to LDS, with HW padding of 4 DWORDs after every 16 DWORDs so each
// 32-half row lands at a 40-half LDS stride (bank-conflict-free for frags).
// D# layout per CDNA5 ISA.
// ---------------------------------------------------------------------------
static __device__ __forceinline__ void tdm_load_tile(const _Float16* gsrc,
                                                     uint32_t lds_bytes,
                                                     uint32_t tensor_rows) {
    uint64_t ga = (uint64_t)(uintptr_t)gsrc;
    u32x4 g0;
    g0[0] = 1u;                                   // count=1, user descriptor
    g0[1] = lds_bytes;                            // lds_addr (bytes)
    g0[2] = (uint32_t)ga;                         // global_addr[31:0]
    g0[3] = (uint32_t)((ga >> 32) & 0x1FFFFFFu)   // global_addr[56:32]
            | (2u << 30);                         // type = 2 ("image")
    i32x8 g1;
    g1[0] = (1 << 16)                             // data_size = 2 bytes
          | (1 << 20)                             // pad_enable
          | (3 << 22)                             // pad_interval: 16 DWORDs
          | (3 << 25);                            // pad_amount: 4 DWORDs
    g1[1] = (int)(1024u << 16);                   // tensor_dim0 = 1024 (lo16)
    g1[2] = (int)((tensor_rows & 0xFFFFu) << 16); // dim0 hi=0 | tensor_dim1 lo
    g1[3] = (int)((tensor_rows >> 16) | (32u << 16)); // dim1 hi | tile_dim0=32
    g1[4] = 128;                                  // tile_dim1=128, tile_dim2=0
    g1[5] = 1024;                                 // tensor_dim0_stride lo
    g1[6] = 0;                                    // stride hi | dim1_stride lo
    g1[7] = 0;
    i32x4 z4 = {0, 0, 0, 0};
    i32x8 z8 = {0, 0, 0, 0, 0, 0, 0, 0};
    __builtin_amdgcn_tensor_load_to_lds(g0, g1, z4, z4, z8, 0);
}

static __device__ __forceinline__ uint32_t lds_off(const void* p) {
    return (uint32_t)(uintptr_t)p;   // low 32 bits of generic addr = LDS offset
}

// ---------------------------------------------------------------------------
// fp32 -> f16 conversion (vectorized 4-wide)
// ---------------------------------------------------------------------------
__global__ void cvt_f32_f16(const float* __restrict__ src,
                            _Float16* __restrict__ dst, int n4) {
    int i = blockIdx.x * blockDim.x + threadIdx.x;
    if (i < n4) {
        f4 v = *(const f4*)&src[i * 4];
        h4 o;
#pragma unroll
        for (int j = 0; j < 4; ++j) o[j] = (_Float16)v[j];
        *(h4*)&dst[i * 4] = o;
    }
}

// ---------------------------------------------------------------------------
// Y[m][n] = sum_k X[m][k] * W[n][k] + bias[n]        (K = N = 1024)
// Block tile 128x128; K-slice 32 staged by TDM into double-buffered LDS.
// 8 waves, each computes a 32x64 sub-tile (8 WMMA accumulators).
// ---------------------------------------------------------------------------
template <typename OutT>
__global__ __launch_bounds__(256) void gemm_xwt_bias(
    const _Float16* __restrict__ X, const _Float16* __restrict__ W,
    const float* __restrict__ bias, OutT* __restrict__ Y) {
    __shared__ __attribute__((aligned(128))) _Float16 ldsX[2][128 * 40];
    __shared__ __attribute__((aligned(128))) _Float16 ldsW[2][128 * 40];

    const int tid  = threadIdx.x;
    const int lane = tid & 31;
    const int wid  = tid >> 5;
    const int l15  = lane & 15;
    const int lhi  = lane >> 4;               // 0 or 1
    const int m0   = blockIdx.y * 128;
    const int n0   = blockIdx.x * 128;
    const int wm   = (wid & 3) * 32;          // wave row offset (4 wave-rows)
    const int wn   = (wid >> 2) * 64;         // wave col offset (2 wave-cols)

    const _Float16* Xt = X + (size_t)m0 * D_MODEL;
    const _Float16* Wt = W + (size_t)n0 * D_MODEL;

    f8 acc[2][4];
#pragma unroll
    for (int t = 0; t < 2; ++t)
#pragma unroll
        for (int u = 0; u < 4; ++u) acc[t][u] = (f8){};

    // prologue: DMA first K-slice into buffer 0 (one wave drives the TDM)
    if (wid == 0) {
        tdm_load_tile(Xt, lds_off(&ldsX[0][0]), MTOT);
        tdm_load_tile(Wt, lds_off(&ldsW[0][0]), D_MODEL);
    }

    for (int kk = 0; kk < D_MODEL; kk += 32) {
        const int buf = (kk >> 5) & 1;
        const int nxt = buf ^ 1;
        if (wid == 0) {
            if (kk + 32 < D_MODEL) {
                // prefetch next slice while this slice is consumed
                tdm_load_tile(Xt + kk + 32, lds_off(&ldsX[nxt][0]), MTOT);
                tdm_load_tile(Wt + kk + 32, lds_off(&ldsW[nxt][0]), D_MODEL);
                __builtin_amdgcn_s_wait_tensorcnt(2);  // current slice landed
            } else {
                __builtin_amdgcn_s_wait_tensorcnt(0);
            }
        }
        __syncthreads();   // release compute on `buf`

        h16 af[2], bf[4];
#pragma unroll
        for (int t = 0; t < 2; ++t) {
            int r = wm + t * 16 + l15;        // A: lane row, kbase=8*lhi, +0/+16
            af[t] = ld2(&ldsX[buf][r * 40 + lhi * 8],
                        &ldsX[buf][r * 40 + lhi * 8 + 16]);
        }
#pragma unroll
        for (int u = 0; u < 4; ++u) {
            int r = wn + u * 16 + l15;        // B: lane col, kbase=8*lhi, +0/+16
            bf[u] = ld2(&ldsW[buf][r * 40 + lhi * 8],
                        &ldsW[buf][r * 40 + lhi * 8 + 16]);
        }
#pragma unroll
        for (int t = 0; t < 2; ++t)
#pragma unroll
            for (int u = 0; u < 4; ++u) acc[t][u] = wmma_f16(af[t], bf[u], acc[t][u]);

        __syncthreads();   // everyone done with `buf` before it is re-filled
    }

    // epilogue: C layout => row m = r + 8*lhi, col n = l15
#pragma unroll
    for (int u = 0; u < 4; ++u) {
        int n = n0 + wn + u * 16 + l15;
        float bv = bias[n];
#pragma unroll
        for (int t = 0; t < 2; ++t)
#pragma unroll
            for (int r = 0; r < 8; ++r) {
                int m = m0 + wm + t * 16 + r + 8 * lhi;
                Y[(size_t)m * D_MODEL + n] = (OutT)(acc[t][u][r] + bv);
            }
    }
}

// ---------------------------------------------------------------------------
// Flash-style attention. grid = (SEQ/128, BATCH*NHEAD), 256 threads.
// Each wave: 16 query rows; loops keys in blocks of 32.
// ---------------------------------------------------------------------------
__global__ __launch_bounds__(256) void attn_kernel(
    const _Float16* __restrict__ Q, const _Float16* __restrict__ K,
    const _Float16* __restrict__ V, _Float16* __restrict__ O) {
    __shared__ _Float16 Vt[64 * 40];          // V transposed: [d][key]
    __shared__ _Float16 Pl[8][16 * 40];       // per-wave P: [q][key]

    const int tid  = threadIdx.x;
    const int lane = tid & 31;
    const int wid  = tid >> 5;
    const int l15  = lane & 15;
    const int lhi  = lane >> 4;
    const int b    = blockIdx.y / NHEAD;
    const int h    = blockIdx.y % NHEAD;
    const int q0   = blockIdx.x * 128 + wid * 16;

    // Q fragments (registers for the whole kernel)
    const size_t qrow = (size_t)(b * SEQ + q0 + l15) * D_MODEL + h * DHEAD;
    h16 qf[2];
#pragma unroll
    for (int df = 0; df < 2; ++df)
        qf[df] = ld2(&Q[qrow + df * 32 + lhi * 8], &Q[qrow + df * 32 + lhi * 8 + 16]);

    f8 o[4];
#pragma unroll
    for (int u = 0; u < 4; ++u) o[u] = (f8){};
    float mrow[8], lrow[8];
#pragma unroll
    for (int r = 0; r < 8; ++r) { mrow[r] = -__builtin_inff(); lrow[r] = 0.0f; }

    const float csc = 0.125f * 1.44269504088896f;   // log2(e)/sqrt(Dh)

    for (int kb = 0; kb < SEQ; kb += 32) {
        __syncthreads();   // previous iteration's LDS consumers done

        // cooperative transpose of V block (32 keys x 64 d) into Vt
        {
            int key = tid >> 3;
            int seg = tid & 7;
            h8 v = *(const h8*)&V[(size_t)(b * SEQ + kb + key) * D_MODEL +
                                  h * DHEAD + seg * 8];
#pragma unroll
            for (int i = 0; i < 8; ++i) Vt[(seg * 8 + i) * 40 + key] = v[i];
        }

        // S = Q * K^T  (16 queries x 32 keys), 2 col-tiles x 2 K-depth frags
        f8 s[2];
        s[0] = (f8){}; s[1] = (f8){};
#pragma unroll
        for (int jj = 0; jj < 2; ++jj) {
            const size_t krow =
                (size_t)(b * SEQ + kb + jj * 16 + l15) * D_MODEL + h * DHEAD;
#pragma unroll
            for (int df = 0; df < 2; ++df) {
                h16 kf = ld2(&K[krow + df * 32 + lhi * 8],
                             &K[krow + df * 32 + lhi * 8 + 16]);
                s[jj] = wmma_f16(qf[df], kf, s[jj]);
            }
        }

        // online softmax (exp2 domain), stats per row r (row = r + 8*lhi)
        float corr[8];
#pragma unroll
        for (int r = 0; r < 8; ++r) {
            float a0 = s[0][r] * csc, a1 = s[1][r] * csc;
            float mx = fmaxf(a0, a1);
            mx = fmaxf(mx, __shfl_xor(mx, 1, 16));
            mx = fmaxf(mx, __shfl_xor(mx, 2, 16));
            mx = fmaxf(mx, __shfl_xor(mx, 4, 16));
            mx = fmaxf(mx, __shfl_xor(mx, 8, 16));
            float mnew = fmaxf(mrow[r], mx);
            corr[r] = exp2f(mrow[r] - mnew);
            mrow[r] = mnew;
            float p0 = exp2f(a0 - mnew), p1 = exp2f(a1 - mnew);
            s[0][r] = p0; s[1][r] = p1;
            float rs = p0 + p1;
            rs += __shfl_xor(rs, 1, 16);
            rs += __shfl_xor(rs, 2, 16);
            rs += __shfl_xor(rs, 4, 16);
            rs += __shfl_xor(rs, 8, 16);
            lrow[r] = lrow[r] * corr[r] + rs;
        }
#pragma unroll
        for (int u = 0; u < 4; ++u)
#pragma unroll
            for (int r = 0; r < 8; ++r) o[u][r] *= corr[r];

        // P (C-layout) -> LDS as f16 so it can be reloaded in A-layout
#pragma unroll
        for (int r = 0; r < 8; ++r) {
            int q = r + 8 * lhi;
            Pl[wid][q * 40 + l15]      = (_Float16)s[0][r];
            Pl[wid][q * 40 + 16 + l15] = (_Float16)s[1][r];
        }
        __syncthreads();   // Vt + P visible

        // O += P (16x32) * V (32x64): 1 A-frag, 4 B-frags
        h16 pf = ld2(&Pl[wid][l15 * 40 + lhi * 8],
                     &Pl[wid][l15 * 40 + lhi * 8 + 16]);
#pragma unroll
        for (int u = 0; u < 4; ++u) {
            h16 vf = ld2(&Vt[(u * 16 + l15) * 40 + lhi * 8],
                         &Vt[(u * 16 + l15) * 40 + lhi * 8 + 16]);
            o[u] = wmma_f16(pf, vf, o[u]);
        }
    }

    // normalize and store (f16, concat-head layout)
#pragma unroll
    for (int r = 0; r < 8; ++r) {
        float inv = 1.0f / lrow[r];
        int m = b * SEQ + q0 + r + 8 * lhi;
#pragma unroll
        for (int u = 0; u < 4; ++u)
            O[(size_t)m * D_MODEL + h * DHEAD + u * 16 + l15] =
                (_Float16)(o[u][r] * inv);
    }
}

// ---------------------------------------------------------------------------
// Host-side launch
// ---------------------------------------------------------------------------
extern "C" void kernel_launch(void* const* d_in, const int* in_sizes, int n_in,
                              void* d_out, int out_size, void* d_ws,
                              size_t ws_size, hipStream_t stream) {
    const float* x_q   = (const float*)d_in[0];
    const float* x_kv  = (const float*)d_in[1];
    const float* W_q   = (const float*)d_in[2];
    const float* b_q   = (const float*)d_in[3];
    const float* W_k   = (const float*)d_in[4];
    const float* b_k   = (const float*)d_in[5];
    const float* W_v   = (const float*)d_in[6];
    const float* b_v   = (const float*)d_in[7];
    const float* W_out = (const float*)d_in[8];
    const float* b_out = (const float*)d_in[9];

    _Float16* ws = (_Float16*)d_ws;
    const size_t XSZ = (size_t)MTOT * D_MODEL;     // 4 Mi halves
    const size_t WSZ = (size_t)D_MODEL * D_MODEL;  // 1 Mi halves
    _Float16* Xq16  = ws;
    _Float16* Xkv16 = Xq16 + XSZ;
    _Float16* Wq16  = Xkv16 + XSZ;
    _Float16* Wk16  = Wq16 + WSZ;
    _Float16* Wv16  = Wk16 + WSZ;
    _Float16* Wo16  = Wv16 + WSZ;
    _Float16* Q16   = Wo16 + WSZ;
    _Float16* K16   = Q16 + XSZ;
    _Float16* V16   = K16 + XSZ;
    _Float16* AO16  = V16 + XSZ;

    const int xn4 = (int)(XSZ / 4), wn4 = (int)(WSZ / 4);
    cvt_f32_f16<<<(xn4 + 255) / 256, 256, 0, stream>>>(x_q, Xq16, xn4);
    cvt_f32_f16<<<(xn4 + 255) / 256, 256, 0, stream>>>(x_kv, Xkv16, xn4);
    cvt_f32_f16<<<(wn4 + 255) / 256, 256, 0, stream>>>(W_q, Wq16, wn4);
    cvt_f32_f16<<<(wn4 + 255) / 256, 256, 0, stream>>>(W_k, Wk16, wn4);
    cvt_f32_f16<<<(wn4 + 255) / 256, 256, 0, stream>>>(W_v, Wv16, wn4);
    cvt_f32_f16<<<(wn4 + 255) / 256, 256, 0, stream>>>(W_out, Wo16, wn4);

    dim3 ggrid(D_MODEL / 128, MTOT / 128);   // (8, 32)
    gemm_xwt_bias<_Float16><<<ggrid, 256, 0, stream>>>(Xq16, Wq16, b_q, Q16);
    gemm_xwt_bias<_Float16><<<ggrid, 256, 0, stream>>>(Xkv16, Wk16, b_k, K16);
    gemm_xwt_bias<_Float16><<<ggrid, 256, 0, stream>>>(Xkv16, Wv16, b_v, V16);

    attn_kernel<<<dim3(SEQ / 128, BATCH * NHEAD), 256, 0, stream>>>(Q16, K16,
                                                                    V16, AO16);

    gemm_xwt_bias<float><<<ggrid, 256, 0, stream>>>(AO16, Wo16, b_out,
                                                    (float*)d_out);
}
